// QwenVL_PartB_20968030339738
// MI455X (gfx1250) — compile-verified, matrix operand order checked by reading, not run on hardware
//
#include <hip/hip_runtime.h>
#include <hip/hip_bf16.h>

// ---------------------------------------------------------------------------
// Types for CDNA5 WMMA (gfx1250, wave32)
// ---------------------------------------------------------------------------
typedef __attribute__((ext_vector_type(16))) __bf16 bf16x16;
typedef __attribute__((ext_vector_type(8)))  __bf16 bf16x8;
typedef __attribute__((ext_vector_type(8)))  float  f32x8;

static __device__ inline bf16x16 frag_combine(bf16x8 lo, bf16x8 hi) {
  return __builtin_shufflevector(lo, hi, 0,1,2,3,4,5,6,7,8,9,10,11,12,13,14,15);
}

static __device__ inline bf16x8 cvt8(float4 a, float4 b) {
  bf16x8 r;
  r[0] = (__bf16)a.x; r[1] = (__bf16)a.y; r[2] = (__bf16)a.z; r[3] = (__bf16)a.w;
  r[4] = (__bf16)b.x; r[5] = (__bf16)b.y; r[6] = (__bf16)b.z; r[7] = (__bf16)b.w;
  return r;
}

static __device__ inline float gelu_tanh(float x) {
  float x3 = x * x * x;
  return 0.5f * x * (1.0f + tanhf(0.7978845608028654f * (x + 0.044715f * x3)));
}

// ---------------------------------------------------------------------------
// Batched WMMA GEMM: per batch z (blockIdx.z):
//   C[z][M,N] = op( A[z][M,K] @ W[z][N,K]^T + bias[N] ) (+ add[z][M,N])
//   REQUIRES: K % 32 == 0 (caller zero-pads K).  lda/ldb multiples of 4 floats.
//   Ragged M/N handled by clamping load rows (results discarded by the
//   guarded epilogue) -> staging is branch-free vector loads.
//   op: 0 = identity, 1 = tanh-gelu.  bias/add may be nullptr.
// 256 threads = 8 wave32; block tile 128(M) x 64(N); K chunk 32 (bf16 WMMA).
// Waves: 8 along M; each wave computes 16x64 = 1 A-frag x 4 B-frags = 4 WMMA.
// ---------------------------------------------------------------------------
#define BM 128
#define BN 64
#define BKC 32

__global__ __launch_bounds__(256) void gemm_wmma_kernel(
    const float* __restrict__ A, int lda, size_t sA,
    const float* __restrict__ W, int ldb, size_t sB,
    const float* __restrict__ bias,
    const float* __restrict__ add,
    float* __restrict__ C, int ldc, size_t sC,
    int M, int N, int K, int op)
{
  __shared__ alignas(16) __bf16 As[BM][BKC];
  __shared__ alignas(16) __bf16 Bs[BN][BKC];

  const int tid  = threadIdx.x;
  const int lane = tid & 31;
  const int wave = tid >> 5;          // 0..7 along M
  const int hl   = lane >> 4;         // half-wave selector
  const int l16  = lane & 15;
  const int m0   = blockIdx.y * BM;
  const int n0   = blockIdx.x * BN;
  const size_t z = blockIdx.z;

  const float* Ab = A + z * sA;
  const float* Wb = W + z * sB;
  float*       Cb = C + z * sC;
  const float* addb = add ? add + z * sC : (const float*)0;

  // cooperative-load mapping (clamped rows: no guards needed in the loop)
  const int arow = tid >> 1;          // 0..127, 16 floats each (two bf16x8)
  const int akb  = (tid & 1) * 16;    // 0 or 16
  const int brow = tid >> 2;          // 0..63, 8 floats each (one bf16x8)
  const int bkb  = (tid & 3) * 8;     // 0,8,16,24

  const int gm = min(m0 + arow, M - 1);
  const int gn = min(n0 + brow, N - 1);
  const float* srcA = Ab + (size_t)gm * lda + akb;
  const float* srcB = Wb + (size_t)gn * ldb + bkb;

  const f32x8 zero8 = {0.f,0.f,0.f,0.f,0.f,0.f,0.f,0.f};
  f32x8 acc[4];
#pragma unroll
  for (int s = 0; s < 4; ++s) acc[s] = zero8;

  for (int k0 = 0; k0 < K; k0 += BKC) {
    // ---- stage A tile: 16 floats/thread -> 2x bf16x8 (global_load_b128 x4) ----
    {
      float4 f0 = *(const float4*)(srcA);
      float4 f1 = *(const float4*)(srcA + 4);
      float4 f2 = *(const float4*)(srcA + 8);
      float4 f3 = *(const float4*)(srcA + 12);
      __builtin_prefetch(srcA + BKC, 0, 3);
      *(bf16x8*)&As[arow][akb]     = cvt8(f0, f1);
      *(bf16x8*)&As[arow][akb + 8] = cvt8(f2, f3);
    }
    // ---- stage B tile: 8 floats/thread -> 1x bf16x8 (global_load_b128 x2) ----
    {
      float4 f0 = *(const float4*)(srcB);
      float4 f1 = *(const float4*)(srcB + 4);
      __builtin_prefetch(srcB + BKC, 0, 3);
      *(bf16x8*)&Bs[brow][bkb] = cvt8(f0, f1);
    }
    srcA += BKC;
    srcB += BKC;
    __syncthreads();

    // ---- fragments per ISA 7.12.2 16-bit layout ----
    // elem 0..7 -> K = hl*8..hl*8+7 ; elem 8..15 -> K = 16+hl*8..
    const int ar = wave * 16 + l16;
    const bf16x16 afrag = frag_combine(*(const bf16x8*)&As[ar][hl * 8],
                                       *(const bf16x8*)&As[ar][16 + hl * 8]);
#pragma unroll
    for (int s = 0; s < 4; ++s) {
      const int br = s * 16 + l16;
      const bf16x16 bfrag = frag_combine(*(const bf16x8*)&Bs[br][hl * 8],
                                         *(const bf16x8*)&Bs[br][16 + hl * 8]);
      acc[s] = __builtin_amdgcn_wmma_f32_16x16x32_bf16(false, afrag, false, bfrag,
                                                       (short)0, acc[s], false, false);
    }
    __syncthreads();
  }

  // ---- epilogue: bias, activation, optional residual, guarded store ----
#pragma unroll
  for (int s = 0; s < 4; ++s) {
    const int col = n0 + s * 16 + l16;
    if (col < N) {
      const float bv = bias ? bias[col] : 0.0f;
#pragma unroll
      for (int r = 0; r < 8; ++r) {
        const int row = m0 + wave * 16 + hl * 8 + r;  // C VGPR r: M=r (+8 upper half)
        if (row < M) {
          float v = acc[s][r] + bv;
          if (op == 1) v = gelu_tanh(v);
          if (addb) v += addb[(size_t)row * ldc + col];
          Cb[(size_t)row * ldc + col] = v;
        }
      }
    }
  }
}

// ---------------------------------------------------------------------------
// LayerNorm over rows of length R (one block per row, wave32 reductions)
// ---------------------------------------------------------------------------
__global__ __launch_bounds__(256) void layernorm_rows(
    const float* __restrict__ X, const float* __restrict__ w,
    const float* __restrict__ b, float* __restrict__ Y, int R)
{
  const int row = blockIdx.x;
  const float* x = X + (size_t)row * R;
  float* y = Y + (size_t)row * R;
  const int tid = threadIdx.x;
  float s = 0.f, s2 = 0.f;
  for (int j = tid; j < R; j += 256) { float v = x[j]; s += v; s2 += v * v; }
#pragma unroll
  for (int off = 16; off > 0; off >>= 1) {
    s  += __shfl_xor(s,  off, 32);
    s2 += __shfl_xor(s2, off, 32);
  }
  __shared__ float rs[8], rs2[8];
  if ((tid & 31) == 0) { rs[tid >> 5] = s; rs2[tid >> 5] = s2; }
  __syncthreads();
  s = 0.f; s2 = 0.f;
#pragma unroll
  for (int i = 0; i < 8; ++i) { s += rs[i]; s2 += rs2[i]; }
  const float mu  = s / (float)R;
  const float var = s2 / (float)R - mu * mu;
  const float inv = rsqrtf(var + 1e-6f);
  for (int j = tid; j < R; j += 256) y[j] = (x[j] - mu) * inv * w[j] + b[j];
}

// ---------------------------------------------------------------------------
// Batched row softmax over n valid columns, then zero-fill [n, ld) so the
// following GEMM can run with K padded to ld. In-place.
// grid: (rows, batch); batch stride sS floats.
// ---------------------------------------------------------------------------
__global__ __launch_bounds__(256) void softmax_rows(
    float* __restrict__ S, int ld, int n, size_t sS)
{
  float* p = S + (size_t)blockIdx.y * sS + (size_t)blockIdx.x * ld;
  const int tid = threadIdx.x;
  float mx = -3.0e38f;
  for (int j = tid; j < n; j += 256) mx = fmaxf(mx, p[j]);
#pragma unroll
  for (int off = 16; off > 0; off >>= 1) mx = fmaxf(mx, __shfl_xor(mx, off, 32));
  __shared__ float rm[8], rsum[8];
  if ((tid & 31) == 0) rm[tid >> 5] = mx;
  __syncthreads();
  mx = rm[0];
#pragma unroll
  for (int i = 1; i < 8; ++i) mx = fmaxf(mx, rm[i]);
  float s = 0.f;
  for (int j = tid; j < n; j += 256) { float e = __expf(p[j] - mx); p[j] = e; s += e; }
#pragma unroll
  for (int off = 16; off > 0; off >>= 1) s += __shfl_xor(s, off, 32);
  if ((tid & 31) == 0) rsum[tid >> 5] = s;
  __syncthreads();
  s = 0.f;
#pragma unroll
  for (int i = 0; i < 8; ++i) s += rsum[i];
  const float inv = 1.0f / s;
  for (int j = tid; j < n; j += 256) p[j] *= inv;
  for (int j = n + tid; j < ld; j += 256) p[j] = 0.0f;   // K padding
}

// ---------------------------------------------------------------------------
// Patchify: pixel_values [1,1,3,480,480] -> xp [900,1536]
// token = ((hf*15+wf)*2+mh)*2+mw ; feature = ((c*2+t)*16+pi)*16+pj
// ---------------------------------------------------------------------------
__global__ __launch_bounds__(256) void patchify(const float* __restrict__ px,
                                                float* __restrict__ xp)
{
  int idx = blockIdx.x * 256 + threadIdx.x;
  if (idx >= 900 * 1536) return;
  int ntok = idx / 1536, f = idx % 1536;
  int mw = ntok & 1, mh = (ntok >> 1) & 1;
  int t2 = ntok >> 2;
  int wf = t2 % 15, hf = t2 / 15;
  int pj = f & 15;
  int r = f >> 4;
  int pi = r & 15; r >>= 4;
  int c = r >> 1;                         // temporal index unused (duplicated)
  int H = (hf * 2 + mh) * 16 + pi;
  int W = (wf * 2 + mw) * 16 + pj;
  float v = px[((size_t)c * 480 + H) * 480 + W];
  xp[idx] = (v - 127.5f) * (1.0f / 127.5f);
}

// ---------------------------------------------------------------------------
// QKV split + RoPE with K-padding:
//   qkv [900,3456] -> qh/kh [h][900][96] (RoPE'd, d=72..95 zeroed)
//                     vT   [h][72][928]  (cols 900..927 zeroed separately)
// ---------------------------------------------------------------------------
#define DHP 96
#define NTP 928

__global__ __launch_bounds__(256) void qkv_rope(
    const float* __restrict__ qkv, const float* __restrict__ cosp,
    const float* __restrict__ sinp, float* __restrict__ qh,
    float* __restrict__ kh, float* __restrict__ vT)
{
  int idx = blockIdx.x * 256 + threadIdx.x;   // over 900 * 16 * 96
  if (idx >= 900 * 16 * DHP) return;
  int n  = idx / (16 * DHP);
  int e  = idx % (16 * DHP);
  int h  = e / DHP, d = e % DHP;
  float* qdst = qh + ((size_t)h * 900 + n) * DHP + d;
  float* kdst = kh + ((size_t)h * 900 + n) * DHP + d;
  if (d >= 72) { *qdst = 0.f; *kdst = 0.f; return; }
  const float* base = qkv + (size_t)n * 3456;
  float q = base[h * 72 + d];
  float k = base[1152 + h * 72 + d];
  float v = base[2304 + h * 72 + d];
  float c = cosp[n * 72 + d];
  float s = sinp[n * 72 + d];
  int dro = (d < 36) ? d + 36 : d - 36;
  float qr = (d < 36) ? -base[h * 72 + dro] : base[h * 72 + dro];
  float kr = (d < 36) ? -base[1152 + h * 72 + dro] : base[1152 + h * 72 + dro];
  *qdst = q * c + qr * s;
  *kdst = k * c + kr * s;
  vT[((size_t)h * 72 + d) * NTP + n] = v;
}

// one-shot: zero vT's padded token columns [900, 928) for all 16*72 rows
__global__ __launch_bounds__(256) void zero_vt_tail(float* __restrict__ vT)
{
  int idx = blockIdx.x * 256 + threadIdx.x;
  if (idx >= 16 * 72 * (NTP - 900)) return;
  int row = idx / (NTP - 900);
  int c   = idx % (NTP - 900);
  vT[(size_t)row * NTP + 900 + c] = 0.0f;
}

// ---------------------------------------------------------------------------
// Host-side orchestration
// ---------------------------------------------------------------------------
static void gemmb(hipStream_t st, const float* A, int lda, size_t sA,
                  const float* W, int ldb, size_t sB,
                  const float* bias, const float* add, float* C, int ldc, size_t sC,
                  int M, int N, int K, int op, int batch)
{
  dim3 g((N + BN - 1) / BN, (M + BM - 1) / BM, batch);
  gemm_wmma_kernel<<<g, dim3(256), 0, st>>>(A, lda, sA, W, ldb, sB, bias, add,
                                            C, ldc, sC, M, N, K, op);
}

static void gemm(hipStream_t st, const float* A, int lda, const float* W, int ldb,
                 const float* bias, const float* add, float* C, int ldc,
                 int M, int N, int K, int op)
{
  gemmb(st, A, lda, 0, W, ldb, 0, bias, add, C, ldc, 0, M, N, K, op, 1);
}

extern "C" void kernel_launch(void* const* d_in, const int* in_sizes, int n_in,
                              void* d_out, int out_size, void* d_ws, size_t ws_size,
                              hipStream_t stream)
{
  const float* px      = (const float*)d_in[0];
  const float* patch_w = (const float*)d_in[1];
  const float* patch_b = (const float*)d_in[2];
  const float* pos_emb = (const float*)d_in[3];
  const float* cosp    = (const float*)d_in[4];
  const float* sinp    = (const float*)d_in[5];
  const float* ln1_w   = (const float*)d_in[6];
  const float* ln1_b   = (const float*)d_in[7];
  const float* qkv_w   = (const float*)d_in[8];
  const float* qkv_b   = (const float*)d_in[9];
  const float* proj_w  = (const float*)d_in[10];
  const float* proj_b  = (const float*)d_in[11];
  const float* ln2_w   = (const float*)d_in[12];
  const float* ln2_b   = (const float*)d_in[13];
  const float* fc1_w   = (const float*)d_in[14];
  const float* fc1_b   = (const float*)d_in[15];
  const float* fc2_w   = (const float*)d_in[16];
  const float* fc2_b   = (const float*)d_in[17];
  const float* dsn_w   = (const float*)d_in[18];
  const float* dsn_b   = (const float*)d_in[19];
  const float* dsf1_w  = (const float*)d_in[20];
  const float* dsf1_b  = (const float*)d_in[21];
  const float* dsf2_w  = (const float*)d_in[22];
  const float* dsf2_b  = (const float*)d_in[23];
  const float* mn_w    = (const float*)d_in[24];
  const float* mn_b    = (const float*)d_in[25];
  const float* mf1_w   = (const float*)d_in[26];
  const float* mf1_b   = (const float*)d_in[27];
  const float* mf2_w   = (const float*)d_in[28];
  const float* mf2_b   = (const float*)d_in[29];
  float* out = (float*)d_out;

  // workspace layout (floats); every offset is a multiple of 16 floats
  float* ws  = (float*)d_ws;
  float* h   = ws;                          // 900*1152
  float* hn  = h   + (size_t)900 * 1152;    // 900*1152 (== 225*4608)
  float* qkv = hn  + (size_t)900 * 1152;    // 900*3456
  float* qh  = qkv + (size_t)900 * 3456;    // 16*900*96  (K padded to 96)
  float* kh  = qh  + (size_t)16 * 900 * DHP;
  float* vT  = kh  + (size_t)16 * 900 * DHP; // 16*72*928 (K padded to 928)
  float* ob  = vT  + (size_t)16 * 72 * NTP;  // 900*1152
  float* big = ob  + (size_t)900 * 1152;     // 900*4608
  float* xp  = big + (size_t)900 * 4608;     // 900*1536
  float* S   = xp  + (size_t)900 * 1536;     // hb * 900*928 (sized from ws_size)

  // choose per-pass head batch for attention scores from available workspace
  const size_t base_floats = (size_t)(S - ws);
  const size_t avail = (ws_size / 4 > base_floats) ? (ws_size / 4 - base_floats) : 0;
  const size_t sS = (size_t)900 * NTP;
  int hb = 1;
  for (int c = 16; c >= 1; c >>= 1) {
    if ((size_t)c * sS <= avail) { hb = c; break; }
  }

  // one-shot padding init for vT (idempotent; qkv_rope never touches the tail)
  zero_vt_tail<<<(16 * 72 * (NTP - 900) + 255) / 256, 256, 0, stream>>>(vT);

  // patch embed: xp = patchify(px); h = xp @ patch_w^T + patch_b + pos_emb
  patchify<<<(900 * 1536 + 255) / 256, 256, 0, stream>>>(px, xp);
  gemm(stream, xp, 1536, patch_w, 1536, patch_b, pos_emb, h, 1152,
       900, 1152, 1536, 0);

  for (int l = 0; l < 6; ++l) {
    layernorm_rows<<<900, 256, 0, stream>>>(h, ln1_w + l * 1152, ln1_b + l * 1152, hn, 1152);
    gemm(stream, hn, 1152, qkv_w + (size_t)l * 3456 * 1152, 1152,
         qkv_b + (size_t)l * 3456, nullptr, qkv, 3456, 900, 3456, 1152, 0);
    qkv_rope<<<(900 * 16 * DHP + 255) / 256, 256, 0, stream>>>(qkv, cosp, sinp, qh, kh, vT);

    // attention: heads batched through blockIdx.z, hb heads per pass
    for (int g = 0; g < 16; g += hb) {
      gemmb(stream, qh + (size_t)g * 900 * DHP, DHP, (size_t)900 * DHP,
            kh + (size_t)g * 900 * DHP, DHP, (size_t)900 * DHP,
            nullptr, nullptr, S, NTP, sS, 900, 900, DHP, 0, hb);
      softmax_rows<<<dim3(900, hb), 256, 0, stream>>>(S, NTP, 900, sS);
      gemmb(stream, S, NTP, sS,
            vT + (size_t)g * 72 * NTP, NTP, (size_t)72 * NTP,
            nullptr, nullptr, ob + (size_t)g * 72, 1152, 72,
            900, 72, NTP, 0, hb);
    }

    gemm(stream, ob, 1152, proj_w + (size_t)l * 1152 * 1152, 1152,
         proj_b + (size_t)l * 1152, h, h, 1152, 900, 1152, 1152, 0);
    layernorm_rows<<<900, 256, 0, stream>>>(h, ln2_w + l * 1152, ln2_b + l * 1152, hn, 1152);
    gemm(stream, hn, 1152, fc1_w + (size_t)l * 4608 * 1152, 1152,
         fc1_b + (size_t)l * 4608, nullptr, big, 4608, 900, 4608, 1152, 1);
    gemm(stream, big, 4608, fc2_w + (size_t)l * 1152 * 4608, 4608,
         fc2_b + (size_t)l * 1152, h, h, 1152, 900, 1152, 4608, 0);

    const int d = (l == 2) ? 0 : ((l == 5) ? 1 : -1);
    if (d >= 0) {
      layernorm_rows<<<225, 256, 0, stream>>>(h, dsn_w + (size_t)d * 4608,
                                              dsn_b + (size_t)d * 4608, hn, 4608);
      gemm(stream, hn, 4608, dsf1_w + (size_t)d * 4608 * 4608, 4608,
           dsf1_b + (size_t)d * 4608, nullptr, big, 4608, 225, 4608, 4608, 1);
      gemm(stream, big, 4608, dsf2_w + (size_t)d * 2048 * 4608, 4608,
           dsf2_b + (size_t)d * 2048, nullptr, out + (size_t)d * 225 * 2048, 2048,
           225, 2048, 4608, 0);
    }
  }

  // merger: LN over EMBED, then view as [225,4608]
  layernorm_rows<<<900, 256, 0, stream>>>(h, mn_w, mn_b, hn, 1152);
  gemm(stream, hn, 4608, mf1_w, 4608, mf1_b, nullptr, big, 4608, 225, 4608, 4608, 1);
  gemm(stream, big, 4608, mf2_w, 4608, mf2_b, nullptr,
       out + (size_t)2 * 225 * 2048, 2048, 225, 2048, 4608, 0);
}
